// graph_constructor_dynamic_89635967467801
// MI455X (gfx1250) — matrix operand order repaired and checked by reading, not compiled, not run on hardware
//
#include <hip/hip_runtime.h>

#define ALPHA 3.0f
#define NN 2048
#define DD 64
#define LL 3
#define BB 8
#define TT 168
#define KK 20

typedef float    v8f  __attribute__((ext_vector_type(8)));
typedef _Float16 v16h __attribute__((ext_vector_type(16)));
typedef _Float16 v8h  __attribute__((ext_vector_type(8)));

union Frag { v16h v; v8h h[2]; };

// Hardware tanh on gfx1250 (V_TANH_F32) when the toolchain exposes it.
__device__ __forceinline__ float tanh_fast(float x) {
#if __has_builtin(__builtin_amdgcn_tanhf)
    return __builtin_amdgcn_tanhf(x);
#else
    return tanhf(x);
#endif
}

__device__ __forceinline__ float relu_(float x) { return x > 0.f ? x : 0.f; }

// A-matrix 16x32 f16 fragment per CDNA5 ISA layout:
// lanes 0-15: row=lane, VGPR0-3 = K k0+0..7 (16B), VGPR4-7 = K k0+16..23 (16B)
// lanes16-31: row=lane-16, K ranges shifted by +8
__device__ __forceinline__ v16h load_fragA(const _Float16* __restrict__ base,
                                           int row16, int lane, int k0) {
    int r = lane & 15;
    int half = (lane >> 4) & 1;
    const char* p = (const char*)(base + (size_t)(row16 + r) * DD);
    Frag f;
    f.h[0] = *(const v8h*)(p + (size_t)(k0 + half * 8) * 2);
    f.h[1] = *(const v8h*)(p + (size_t)(k0 + 16 + half * 8) * 2);
    return f.v;
}

// B-matrix 32x16 f16 fragment (B[k][n] = src[J+n][k], so column n is a
// contiguous source row): lanes 0-15: col=lane, K k0+0..7 and k0+16..23;
// lanes 16-31: col=lane-16, K ranges shifted by +8
__device__ __forceinline__ v16h load_fragB(const _Float16* __restrict__ base,
                                           int row16, int lane, int k0) {
    int r = lane & 15;
    int half = (lane >> 4) & 1;
    const char* p = (const char*)(base + (size_t)(row16 + r) * DD);
    Frag f;
    f.h[0] = *(const v8h*)(p + (size_t)(k0 + half * 8) * 2);
    f.h[1] = *(const v8h*)(p + (size_t)(k0 + 16 + half * 8) * 2);
    return f.v;
}

// -------- stage 0a: var_relu[b][d][n] = relu(x[b,n,:]·W1_w[d,:] + W1_b[d])
__global__ void var_kernel(const float* __restrict__ x, const float* __restrict__ W1_w,
                           const float* __restrict__ W1_b, float* __restrict__ vr) {
    int t = blockIdx.x * blockDim.x + threadIdx.x;      // B*D*N = 2^20
    int n = t & (NN - 1);
    int d = (t >> 11) & (DD - 1);
    int b = t >> 17;
    const float* xr = x + ((size_t)b * NN + n) * TT;
    const float* w  = W1_w + (size_t)d * TT;
    float s = W1_b[d];
    for (int i = 0; i < TT; ++i) s += xr[i] * w[i];
    vr[t] = relu_(s);
}

// -------- stage 0b: DI[b][d][e] = relu(sum_n vr[b,d,n]*W2_w[e,n] + W2_b[e])
__global__ void di_kernel(const float* __restrict__ vr, const float* __restrict__ W2_w,
                          const float* __restrict__ W2_b, float* __restrict__ DI) {
    int t = blockIdx.x * blockDim.x + threadIdx.x;      // B*D*D = 32768
    int e = t & (DD - 1);
    int d = (t >> 6) & (DD - 1);
    int b = t >> 12;
    const float* v = vr + ((size_t)b * DD + d) * NN;
    const float* w = W2_w + (size_t)e * NN;
    float s = W2_b[e];
    for (int n = 0; n < NN; ++n) s += v[n] * w[n];
    DI[t] = relu_(s);
}

// -------- stage 0c: nv[j][n][d] = emb[j][idx[n]][d]
__global__ void gather_kernel(const float* __restrict__ emb, const int* __restrict__ idx,
                              float* __restrict__ nv) {
    int t = blockIdx.x * blockDim.x + threadIdx.x;      // 4*N*D = 2^19
    int d = t & (DD - 1);
    int n = (t >> 6) & (NN - 1);
    int j = t >> 17;
    nv[t] = emb[((size_t)j * NN + idx[n]) * DD + d];
}

// -------- per-layer: nv_out[j,n,e] = tanh(A*(scale*nv_in[j,n,:]·w[j,i,e,:] + b[j,i,e]))
__global__ void nv_update_kernel(const float* __restrict__ nv_in,
                                 const float* __restrict__ lin_w,
                                 const float* __restrict__ lin_b,
                                 const float* __restrict__ scale_set, int layer,
                                 float* __restrict__ nv_out,
                                 _Float16* __restrict__ nv0h, _Float16* __restrict__ nv1h) {
    int t = blockIdx.x * blockDim.x + threadIdx.x;      // 4*N*D = 2^19
    int e = t & (DD - 1);
    int n = (t >> 6) & (NN - 1);
    int j = t >> 17;
    const float* w  = lin_w + ((size_t)(j * LL + layer) * DD + e) * DD;
    const float* xr = nv_in + ((size_t)j * NN + n) * DD;
    float s = 0.f;
    for (int d = 0; d < DD; ++d) s += xr[d] * w[d];
    float val = tanh_fast(ALPHA * (s * scale_set[layer] + lin_b[(j * LL + layer) * DD + e]));
    nv_out[t] = val;
    if (j == 0)      nv0h[n * DD + e] = (_Float16)val;
    else if (j == 1) nv1h[n * DD + e] = (_Float16)val;
}

// -------- per-layer: MD{1,2}[b,n,e] = f16(tanh(A * nv[2|3][n,:]·DI[b,:,e]))
__global__ void md_kernel(const float* __restrict__ nv, const float* __restrict__ DI,
                          _Float16* __restrict__ MD1h, _Float16* __restrict__ MD2h) {
    int t = blockIdx.x * blockDim.x + threadIdx.x;      // 2*B*N*D = 2^21
    int e = t & (DD - 1);
    int n = (t >> 6) & (NN - 1);
    int b = (t >> 17) & (BB - 1);
    int which = t >> 20;
    const float* nr = nv + ((size_t)(2 + which) * NN + n) * DD;
    const float* di = DI + (size_t)b * DD * DD;          // [d][e]
    float s = 0.f;
    for (int d = 0; d < DD; ++d) s += nr[d] * di[d * DD + e];
    _Float16 v = (_Float16)tanh_fast(ALPHA * s);
    (which ? MD2h : MD1h)[((size_t)b * NN + n) * DD + e] = v;
}

// -------- fused WMMA kernel: one wave per 16x32 output tile of adj
// adj[n,m] = (1/8) * sum_b relu(tanh( S[n,m] + relu(tanh(A*(a1-a1^T))) ))
// a1 - a1^T at (I,J) = MD1_I·MD2_J^T - MD2_I·MD1_J^T  (no lane transposes!)
// Two J-subtiles per wave so the four I-side A-fragments are reused.
__global__ __launch_bounds__(256) void adj_wmma_kernel(
    const _Float16* __restrict__ nv0h, const _Float16* __restrict__ nv1h,
    const _Float16* __restrict__ MD1h, const _Float16* __restrict__ MD2h,
    float* __restrict__ outAdj) {
    int lane = threadIdx.x & 31;
    int wave = threadIdx.x >> 5;
    int tile = blockIdx.x * 8 + wave;                    // 0..8191
    int I  = (tile >> 6) * 16;                           // 128 row-tiles
    int Jb = (tile & 63) * 32;                           // 64 col-tiles of width 32

    v8f z = {0.f, 0.f, 0.f, 0.f, 0.f, 0.f, 0.f, 0.f};
    v8f S[2], acc[2];

    // static antisymmetric term: nv0_I·nv1_J^T - nv1_I·nv0_J^T
    v16h sa0l = load_fragA(nv0h, I, lane, 0), sa0h = load_fragA(nv0h, I, lane, 32);
    v16h sa1l = load_fragA(nv1h, I, lane, 0), sa1h = load_fragA(nv1h, I, lane, 32);
#pragma unroll
    for (int jt = 0; jt < 2; ++jt) {
        int J = Jb + jt * 16;
        v16h b0l = load_fragB(nv1h, J, lane, 0), b0h = load_fragB(nv1h, J, lane, 32);
        v16h b1l = load_fragB(nv0h, J, lane, 0), b1h = load_fragB(nv0h, J, lane, 32);
        v8f p = __builtin_amdgcn_wmma_f32_16x16x32_f16(false, sa0l, false, b0l, (short)0, z, false, false);
        p     = __builtin_amdgcn_wmma_f32_16x16x32_f16(false, sa0h, false, b0h, (short)0, p, false, false);
        v8f q = __builtin_amdgcn_wmma_f32_16x16x32_f16(false, sa1l, false, b1l, (short)0, z, false, false);
        q     = __builtin_amdgcn_wmma_f32_16x16x32_f16(false, sa1h, false, b1h, (short)0, q, false, false);
#pragma unroll
        for (int r = 0; r < 8; ++r) S[jt][r] = relu_(tanh_fast(ALPHA * (p[r] - q[r])));
        acc[jt] = z;
    }

    for (int b = 0; b < BB; ++b) {
        const _Float16* m1 = MD1h + (size_t)b * NN * DD;
        const _Float16* m2 = MD2h + (size_t)b * NN * DD;
        // I-side A fragments, reused across both J-subtiles
        v16h Al = load_fragA(m1, I, lane, 0), Ah = load_fragA(m1, I, lane, 32);
        v16h Cl = load_fragA(m2, I, lane, 0), Ch = load_fragA(m2, I, lane, 32);
#pragma unroll
        for (int jt = 0; jt < 2; ++jt) {
            int J = Jb + jt * 16;
            v16h Bl = load_fragB(m2, J, lane, 0), Bh = load_fragB(m2, J, lane, 32);
            v16h Dl = load_fragB(m1, J, lane, 0), Dh = load_fragB(m1, J, lane, 32);
            v8f d1 = __builtin_amdgcn_wmma_f32_16x16x32_f16(false, Al, false, Bl, (short)0, z,  false, false);
            d1     = __builtin_amdgcn_wmma_f32_16x16x32_f16(false, Ah, false, Bh, (short)0, d1, false, false);
            v8f d2 = __builtin_amdgcn_wmma_f32_16x16x32_f16(false, Cl, false, Dl, (short)0, z,  false, false);
            d2     = __builtin_amdgcn_wmma_f32_16x16x32_f16(false, Ch, false, Dh, (short)0, d2, false, false);
#pragma unroll
            for (int r = 0; r < 8; ++r) {
                float dy = relu_(tanh_fast(ALPHA * (d1[r] - d2[r])));
                acc[jt][r] += relu_(tanh_fast(S[jt][r] + dy));
            }
        }
    }

    // C/D layout: VGPR r, lanes 0-15 -> M=r, N=lane; lanes 16-31 -> M=r+8
    int col   = lane & 15;
    int rbase = I + ((lane >> 4) << 3);
#pragma unroll
    for (int jt = 0; jt < 2; ++jt)
#pragma unroll
        for (int r = 0; r < 8; ++r)
            outAdj[(size_t)(rbase + r) * NN + Jb + jt * 16 + col] = acc[jt][r] * 0.125f;
}

// -------- in-place top-k(=20) mask per row, emulating jax.lax.top_k ties (lowest index)
__global__ __launch_bounds__(256) void topk_kernel(float* __restrict__ out) {
    __shared__ float vals[NN];
    __shared__ float bv[256];
    __shared__ int   bi[256];
    __shared__ int   sel[KK];
    __shared__ float selv[KK];
    int row = blockIdx.x;
    float* rp = out + (size_t)row * NN;
    int tid = threadIdx.x;
    for (int m = tid; m < NN; m += 256) vals[m] = rp[m];
    __syncthreads();
    for (int it = 0; it < KK; ++it) {
        float best = -1e30f;
        int besti = 0x7fffffff;
        for (int m = tid; m < NN; m += 256) {
            float v = vals[m];
            if (v > best) { best = v; besti = m; }     // strict > keeps lowest index
        }
        bv[tid] = best; bi[tid] = besti;
        __syncthreads();
        for (int s = 128; s > 0; s >>= 1) {
            if (tid < s) {
                float v2 = bv[tid + s]; int i2 = bi[tid + s];
                if (v2 > bv[tid] || (v2 == bv[tid] && i2 < bi[tid])) { bv[tid] = v2; bi[tid] = i2; }
            }
            __syncthreads();
        }
        if (tid == 0) { sel[it] = bi[0]; selv[it] = bv[0]; vals[bi[0]] = -1e30f; }
        __syncthreads();
    }
    for (int m = tid; m < NN; m += 256) vals[m] = 0.f;
    __syncthreads();
    if (tid < KK) vals[sel[tid]] = selv[tid];
    __syncthreads();
    for (int m = tid; m < NN; m += 256) rp[m] = vals[m];
}

extern "C" void kernel_launch(void* const* d_in, const int* in_sizes, int n_in,
                              void* d_out, int out_size, void* d_ws, size_t ws_size,
                              hipStream_t stream) {
    const int*   idx       = (const int*)d_in[0];
    const float* scale_set = (const float*)d_in[1];
    const float* x         = (const float*)d_in[2];
    const float* emb       = (const float*)d_in[3];
    const float* lin_w     = (const float*)d_in[4];
    const float* lin_b     = (const float*)d_in[5];
    const float* W1_w      = (const float*)d_in[6];
    const float* W1_b      = (const float*)d_in[7];
    const float* W2_w      = (const float*)d_in[8];
    const float* W2_b      = (const float*)d_in[9];
    float* out = (float*)d_out;

    char* ws = (char*)d_ws;
    size_t off = 0;
    auto alloc = [&](size_t bytes) { char* p = ws + off; off += (bytes + 255) & ~(size_t)255; return p; };
    float*    vr   = (float*)alloc((size_t)BB * DD * NN * 4);   // 4 MB
    float*    DI   = (float*)alloc((size_t)BB * DD * DD * 4);   // 128 KB
    float*    nvA  = (float*)alloc((size_t)4 * NN * DD * 4);    // 2 MB
    float*    nvB  = (float*)alloc((size_t)4 * NN * DD * 4);    // 2 MB
    _Float16* nv0h = (_Float16*)alloc((size_t)NN * DD * 2);     // 256 KB
    _Float16* nv1h = (_Float16*)alloc((size_t)NN * DD * 2);     // 256 KB
    _Float16* MD1h = (_Float16*)alloc((size_t)BB * NN * DD * 2);// 2 MB
    _Float16* MD2h = (_Float16*)alloc((size_t)BB * NN * DD * 2);// 2 MB
    (void)ws_size; (void)n_in; (void)in_sizes; (void)out_size;

    var_kernel<<<(BB * DD * NN) / 256, 256, 0, stream>>>(x, W1_w, W1_b, vr);
    di_kernel<<<(BB * DD * DD) / 256, 256, 0, stream>>>(vr, W2_w, W2_b, DI);
    gather_kernel<<<(4 * NN * DD) / 256, 256, 0, stream>>>(emb, idx, nvA);

    float* nv_cur = nvA;
    float* nv_nxt = nvB;
    for (int layer = 0; layer < LL; ++layer) {
        nv_update_kernel<<<(4 * NN * DD) / 256, 256, 0, stream>>>(
            nv_cur, lin_w, lin_b, scale_set, layer, nv_nxt, nv0h, nv1h);
        md_kernel<<<(2 * BB * NN * DD) / 256, 256, 0, stream>>>(nv_nxt, DI, MD1h, MD2h);
        adj_wmma_kernel<<<(NN / 16) * (NN / 32) / 8, 256, 0, stream>>>(
            nv0h, nv1h, MD1h, MD2h, out + (size_t)layer * NN * NN);
        float* t = nv_cur; nv_cur = nv_nxt; nv_nxt = t;
    }

    topk_kernel<<<LL * NN, 256, 0, stream>>>(out);
}
